// Multiplication_GRU_40364102647870
// MI455X (gfx1250) — matrix operand order repaired and checked by reading, not compiled, not run on hardware
//
#include <hip/hip_runtime.h>
#include <hip/hip_bf16.h>

typedef __attribute__((ext_vector_type(16))) _Float16 v16h;
typedef __attribute__((ext_vector_type(8)))  _Float16 v8h;
typedef __attribute__((ext_vector_type(8)))  float    v8f;

// Problem constants (from the reference)
constexpr int kB  = 256;   // batch
constexpr int kT  = 2048;  // timesteps
constexpr int kI  = 2;     // input features
constexpr int kH  = 128;   // hidden
constexpr int kBC = 16;    // batch rows per workgroup (WMMA M tile)
constexpr int kNT = 256;   // threads per workgroup = 8 waves (wave32)

// ---- fast transcendentals on the TRANS pipe (v_exp_f32 / v_rcp_f32 / v_tanh_f32) ----
__device__ __forceinline__ float fast_sigmoid(float v) {
    // 1/(1+e^-v) = rcp(1 + exp2(-v*log2e))
    const float e = __builtin_amdgcn_exp2f(v * -1.4426950408889634f);
    return __builtin_amdgcn_rcpf(1.0f + e);
}

__device__ __forceinline__ float fast_tanh(float v) {
#if __has_builtin(__builtin_amdgcn_tanhf)
    return __builtin_amdgcn_tanhf(v);     // v_tanh_f32
#else
    const float e = __builtin_amdgcn_exp2f(v * 2.8853900817779268f);
    return 1.0f - 2.0f * __builtin_amdgcn_rcpf(e + 1.0f);
#endif
}

__global__ __launch_bounds__(kNT)
void gru_scan_wmma_kernel(const float* __restrict__ x,        // [B,T,2]
                          const int*   __restrict__ lengths,  // [B]
                          const float* __restrict__ W_ih,     // [384,2]
                          const float* __restrict__ W_hh,     // [384,128]
                          const float* __restrict__ b_ih,     // [384]
                          const float* __restrict__ b_hh,     // [384]
                          const float* __restrict__ head_w,   // [128]
                          const float* __restrict__ head_b,   // [1]
                          float*       __restrict__ out)      // [B]
{
    // Double-buffered hidden state (f16 for WMMA A fragments), 2*16*128*2B = 8KB
    __shared__ _Float16 hbuf[2][kBC * kH];
    // Last-valid hidden state per row (f32), written once after the scan
    __shared__ float    hlast[kBC * kH];
    // Double-buffered x staging for the current/next timestep
    __shared__ float    xbuf[2][kBC][2];
    __shared__ float    hw[kH];
    __shared__ int      lenb[kBC];

    const int tid   = threadIdx.x;
    const int lane  = tid & 31;
    const int wave  = tid >> 5;          // 0..7
    const int bbase = blockIdx.x * kBC;  // global batch row base

    // ---------------- one-time setup ----------------
    for (int i = tid; i < kBC * kH; i += kNT)
        hbuf[0][i] = (_Float16)0.0f;     // h0 = 0
    if (tid < kH)  hw[tid] = head_w[tid];
    if (tid < kBC) {
        lenb[tid] = lengths[bbase + tid];
        const float* xp = &x[(size_t)(bbase + tid) * kT * kI];
        xbuf[0][tid][0] = xp[0];
        xbuf[0][tid][1] = xp[1];
    }

    const int n  = lane & 15;            // tile column / A row (M)
    const int hl = lane >> 4;            // half-wave select
    const int j  = wave * 16 + n;        // hidden column owned for gate math

    // Per-lane gate coefficients for gate rows j (r), j+128 (z), j+256 (n)
    float wi0[3], wi1[3], bi[3], bh[3];
#pragma unroll
    for (int g = 0; g < 3; ++g) {
        const int gc = j + g * kH;
        wi0[g] = W_ih[gc * kI + 0];
        wi1[g] = W_ih[gc * kI + 1];
        bi[g]  = b_ih[gc];
        bh[g]  = b_hh[gc];
    }

    // Register-resident WMMA B fragments of W_hh^T.
    // Wave w owns column tiles {w, w+8, w+16} (r/z/n gates for the same cols).
    // B layout (32x16 KxN, f16): elems 0-7 hold K=8h..8h+7, elems 8-15 K=16+8h..16+8h+7 (h=lane>>4),
    // N = lane&15.  B[k][nn] = W_hh[nn_global, k]  (since gh = h @ W_hh^T).
    v16h bf[3][4];
#pragma unroll
    for (int g = 0; g < 3; ++g) {
        const int gc = (wave + g * 8) * 16 + n;   // gate row in W_hh
#pragma unroll
        for (int kt = 0; kt < 4; ++kt) {
#pragma unroll
            for (int i = 0; i < 16; ++i)
                bf[g][kt][i] = (_Float16)W_hh[gc * kH + kt * 32 + hl * 8 + (i & 7) + ((i >> 3) << 4)];
        }
    }

    __syncthreads();

    const int m0 = hl * 8;   // D-tile row base for this half-wave
    const int kb = hl * 8;   // A-fragment K base (ISA 16-bit A layout)
    int cur = 0;

    // Registers per lane: current h, captured last-valid h, and len-1 for the
    // 8 (row, col j) elements this lane owns.  Branch-free capture via cndmask.
    float hreg[8], hcap[8];
    int   lenm1[8];
#pragma unroll
    for (int i = 0; i < 8; ++i) {
        hreg[i]  = 0.0f;
        hcap[i]  = 0.0f;
        lenm1[i] = lenb[m0 + i] - 1;
    }

    // ---------------- serial scan over T ----------------
    for (int t = 0; t < kT; ++t) {
        const int nxt = cur ^ 1;

        // Prefetch x for step t+1 (lands in LDS at end of this iteration).
        float nx0 = 0.0f, nx1 = 0.0f;
        if (tid < kBC && (t + 1) < kT) {
            const float* xp = &x[((size_t)(bbase + tid) * kT + (t + 1)) * kI];
            nx0 = xp[0];
            nx1 = xp[1];
        }
        if (tid < kBC && (t + 9) < kT)
            __builtin_prefetch(&x[((size_t)(bbase + tid) * kT + (t + 9)) * kI], 0, 0);

        // A fragments (16x32 f16 of h) from hbuf[cur].
        // Lane row M = lane&15; halves 0-7 -> K=kb..kb+7, halves 8-15 -> K=kb+16..kb+23.
        v16h a[4];
#pragma unroll
        for (int kt = 0; kt < 4; ++kt) {
            const v8h lo = *(const v8h*)&hbuf[cur][n * kH + kt * 32 + kb];
            const v8h hi = *(const v8h*)&hbuf[cur][n * kH + kt * 32 + kb + 16];
#pragma unroll
            for (int i = 0; i < 8; ++i) { a[kt][i] = lo[i]; a[kt][8 + i] = hi[i]; }
        }

        // gh tiles for r/z/n:  D = A x B + C  chained over K = 128.
        v8f acc[3] = {};
#pragma unroll
        for (int kt = 0; kt < 4; ++kt) {
#pragma unroll
            for (int g = 0; g < 3; ++g) {
                acc[g] = __builtin_amdgcn_wmma_f32_16x16x32_f16(
                    false, a[kt], false, bf[g][kt], (short)0, acc[g],
                    false, false);
            }
        }

        // Wave-local elementwise GRU gates + state update (branch-free TRANS math).
#pragma unroll
        for (int i = 0; i < 8; ++i) {
            const int m  = m0 + i;                       // row within batch tile
            const float x0 = xbuf[cur][m][0];
            const float x1 = xbuf[cur][m][1];

            const float xr = fmaf(x1, wi1[0], fmaf(x0, wi0[0], bi[0]));
            const float xz = fmaf(x1, wi1[1], fmaf(x0, wi0[1], bi[1]));
            const float xn = fmaf(x1, wi1[2], fmaf(x0, wi0[2], bi[2]));

            const float hr = acc[0][i] + bh[0];
            const float hz = acc[1][i] + bh[1];
            const float hn = acc[2][i] + bh[2];

            const float r  = fast_sigmoid(xr + hr);
            const float z  = fast_sigmoid(xz + hz);
            const float nn = fast_tanh(fmaf(r, hn, xn));

            const float hv = fmaf(z, hreg[i] - nn, nn);  // (1-z)*n + z*h
            hreg[i] = hv;

            // branch-free last-valid capture (v_cmp + v_cndmask, stays in VGPRs)
            hcap[i] = (t == lenm1[i]) ? hv : hcap[i];

            hbuf[nxt][m * kH + j] = (_Float16)hv;
        }

        // Stage prefetched x for the next step.
        if (tid < kBC && (t + 1) < kT) {
            xbuf[nxt][tid][0] = nx0;
            xbuf[nxt][tid][1] = nx1;
        }

        __syncthreads();   // one barrier per step (double buffering)
        cur = nxt;
    }

    // ---------------- publish captured last states, then head ----------------
#pragma unroll
    for (int i = 0; i < 8; ++i)
        hlast[(m0 + i) * kH + j] = hcap[i];
    __syncthreads();

    // out[b] = hlast[b] . head_w + head_b
    if (tid < kBC) {
        float s = head_b[0];
#pragma unroll 8
        for (int c = 0; c < kH; ++c)
            s = fmaf(hlast[tid * kH + c], hw[c], s);
        out[bbase + tid] = s;
    }
}

extern "C" void kernel_launch(void* const* d_in, const int* in_sizes, int n_in,
                              void* d_out, int out_size, void* d_ws, size_t ws_size,
                              hipStream_t stream) {
    (void)in_sizes; (void)n_in; (void)d_ws; (void)ws_size; (void)out_size;
    const float* x       = (const float*)d_in[0];
    const int*   lengths = (const int*)  d_in[1];
    const float* W_ih    = (const float*)d_in[2];
    const float* W_hh    = (const float*)d_in[3];
    const float* b_ih    = (const float*)d_in[4];
    const float* b_hh    = (const float*)d_in[5];
    const float* head_w  = (const float*)d_in[6];
    const float* head_b  = (const float*)d_in[7];
    float*       out     = (float*)d_out;

    gru_scan_wmma_kernel<<<kB / kBC, kNT, 0, stream>>>(
        x, lengths, W_ih, W_hh, b_ih, b_hh, head_w, head_b, out);
}
